// Molecule_GNN_36009005810072
// MI455X (gfx1250) — matrix operand order repaired and checked by reading, not compiled, not run on hardware
//
#include <hip/hip_runtime.h>
#include <hip/hip_bf16.h>

// ---------------------------------------------------------------------------
// Problem constants (from the reference)
// ---------------------------------------------------------------------------
#define N_NODES   100000
#define E_EDGES   400000
#define ET_EDGES  (E_EDGES + N_NODES)   // with self loops = 500000
#define G_GRAPHS  4000
#define F_IN      32
#define HID       128
#define NHEAD     3
#define W3        384                   // NHEAD * HID
#define NCLS      10
#define NLAYER    4
#define NEG_SLOPE 0.2f

typedef __attribute__((ext_vector_type(16))) __bf16 v16bf;
typedef __attribute__((ext_vector_type(2)))  __bf16 v2bf;
typedef __attribute__((ext_vector_type(8)))  float  v8f;

// ---------------------------------------------------------------------------
// Helpers
// ---------------------------------------------------------------------------
__device__ __forceinline__ unsigned bf_bits(float f) {      // rne f32 -> bf16 bits
    unsigned u = __builtin_bit_cast(unsigned, f);
    return (u + 0x7FFFu + ((u >> 16) & 1u)) >> 16;
}
__device__ __forceinline__ __bf16 f2bf(float f) {
    unsigned short hs = (unsigned short)bf_bits(f);
    return __builtin_bit_cast(__bf16, hs);
}
// pack 4 floats -> 4 bf16 (8 bytes); native v_cvt_pk_bf16_f32 when available
__device__ __forceinline__ uint2 pack4_bf16(float4 v) {
    uint2 r;
#if __has_builtin(__builtin_amdgcn_cvt_pk_bf16_f32)
    v2bf p0 = __builtin_amdgcn_cvt_pk_bf16_f32(v.x, v.y);
    v2bf p1 = __builtin_amdgcn_cvt_pk_bf16_f32(v.z, v.w);
    r.x = __builtin_bit_cast(unsigned, p0);
    r.y = __builtin_bit_cast(unsigned, p1);
#else
    r.x = bf_bits(v.x) | (bf_bits(v.y) << 16);
    r.y = bf_bits(v.z) | (bf_bits(v.w) << 16);
#endif
    return r;
}

// order-preserving float -> uint key (for atomic max over signed floats)
__device__ __forceinline__ unsigned enc_f32(float f) {
    unsigned u = __builtin_bit_cast(unsigned, f);
    return (u & 0x80000000u) ? ~u : (u | 0x80000000u);
}
__device__ __forceinline__ float dec_f32(unsigned k) {
    unsigned u = (k & 0x80000000u) ? (k ^ 0x80000000u) : ~k;
    return __builtin_bit_cast(float, u);
}
#define NEG_INF_KEY 0x007FFFFFu   // enc_f32(-inf)

__device__ __forceinline__ void edge_sd(const int* __restrict__ ei, int e,
                                        int& s, int& d) {
    if (e < E_EDGES) { s = ei[e]; d = ei[E_EDGES + e]; }
    else             { s = d = e - E_EDGES; }           // self loop
}

// ---------------------------------------------------------------------------
// Weight pre-pack: fp32 [K, Nout] row-major -> bf16 fragment-order
//   Wp[((nblk*(K/32) + kblk)*32 + lane)*16 + slot]
// B 16-bit 32x16 fragment, contiguous k order used here: VGPR v holds K = 2v + half*16
//   => lane = (n%16) + 16*(kk>>4), slot = kk&15    (kk = k % 32)
// Per-lane fragment = 16 contiguous bf16 = 32 bytes -> 2x b128 loads.
// ---------------------------------------------------------------------------
__global__ void k_pack_w(const float* __restrict__ W, __bf16* __restrict__ Wp,
                         int K, int Nout)
{
    int i = blockIdx.x * blockDim.x + threadIdx.x;
    if (i >= K * Nout) return;
    int k = i / Nout, n = i % Nout;
    int nblk = n >> 4, kblk = k >> 5, kk = k & 31;
    int lane = (n & 15) + ((kk >> 4) << 4);
    int slot = kk & 15;
    size_t idx = (((size_t)nblk * (K >> 5) + kblk) * 32 + lane) * 16 + slot;
    Wp[idx] = f2bf(W[i]);
}

// ---------------------------------------------------------------------------
// WMMA bf16 GEMM:  C[M,Nout] = act(A[M,K] @ B[K,Nout] + bias)
//   block = 256 threads (8 wave32), tile 128x128, each wave -> 16x128 strip.
//   A staged via LDS in fragment order (per-lane contiguous 32B chunks);
//   B fragments preloaded from pre-packed global (16x global_load_b128, one
//   base + immediate offsets), then an 8-WMMA chain drains them so load
//   latency overlaps matrix execution.
// A 16-bit 16x32 fragment, same contiguous k order: VGPR v holds K = 2v + half*16
//   => lane = m + 16*(k>>4), slot = k&15
//   A quad of consecutive k (k%4==0) -> 4 consecutive slots -> one b64 store.
// ---------------------------------------------------------------------------
#define MT 128
#define NT 128
#define KT 32
#define NTILES (NT / 16)     // 8

__global__ __launch_bounds__(256)
void k_gemm_bf16(const float* __restrict__ A, const __bf16* __restrict__ Bp,
                 const float* __restrict__ bias, float* __restrict__ C,
                 int M, int K, int Nout, int relu)
{
    __shared__ __align__(32) __bf16 As[8][32][16];   // [strip][lane][slot] 8KB

    const int lane = threadIdx.x & 31;
    const int wave = threadIdx.x >> 5;          // 0..7 (= M strip)
    const int half = lane >> 4;                 // 0/1
    const int l16  = lane & 15;
    const int m0   = blockIdx.x * MT;
    const int n0   = blockIdx.y * NT;
    const int nblk0 = n0 >> 4;
    const int kblks = K >> 5;

    v8f acc[NTILES];
#pragma unroll
    for (int t = 0; t < NTILES; ++t)
#pragma unroll
        for (int q = 0; q < 8; ++q) acc[t][q] = 0.0f;

    for (int kb = 0; kb < kblks; ++kb) {
        // ---- stage A tile (128x32): float4 coalesced loads, bf16 quad stores
        //      1024 float4 / 256 threads = 4 per thread
        for (int i = threadIdx.x; i < (MT * KT) / 4; i += 256) {
            int r  = i >> 3;                 // 8 float4 per 32-wide row
            int c4 = (i & 7) << 2;
            int gr = m0 + r;
            float4 v = make_float4(0.f, 0.f, 0.f, 0.f);
            if (gr < M) {
                const float* ap = &A[(size_t)gr * K + (kb << 5) + c4];
                v = *(const float4*)ap;
                if (kb + 1 < kblks)
                    __builtin_prefetch(ap + KT, 0, 1);   // global_prefetch_b8
            }
            uint2 st = pack4_bf16(v);
            int strip = r >> 4, mm = r & 15;
            int ln = mm + ((c4 >> 4) << 4);
            int sb = c4 & 15;   // slot base (mult of 4)
            *(uint2*)&As[strip][ln][sb] = st;
        }
        __syncthreads();

        // ---- preload all B fragments (16x global_load_b128, shared base) ----
        const __bf16* bbase =
            Bp + (((size_t)nblk0 * kblks + kb) * 32 + lane) * 16;
        v16bf bfrag[NTILES];
#pragma unroll
        for (int nt = 0; nt < NTILES; ++nt)
            bfrag[nt] = *(const v16bf*)(bbase + (size_t)nt * kblks * 512);

        // ---- A fragment: 2x ds_load_b128, reused for all ntiles ----
        v16bf af = *(const v16bf*)&As[wave][lane][0];

        // ---- 8-WMMA chain; waits drain the preloaded B in issue order ----
#pragma unroll
        for (int nt = 0; nt < NTILES; ++nt)
            acc[nt] = __builtin_amdgcn_wmma_f32_16x16x32_bf16(
                false, af, false, bfrag[nt], (short)0, acc[nt], false, false);
        __syncthreads();
    }

    // ---- store: C/D layout: VGPR r -> M = r + half*8, N = lane%16 ----
#pragma unroll
    for (int nt = 0; nt < NTILES; ++nt) {
        int col = n0 + nt * 16 + l16;
#pragma unroll
        for (int r = 0; r < 8; ++r) {
            int row = m0 + wave * 16 + half * 8 + r;
            if (row < M) {
                float v = acc[nt][r];
                if (bias) v += bias[col];
                if (relu) v = fmaxf(v, 0.0f);
                C[(size_t)row * Nout + col] = v;
            }
        }
    }
}

// ---------------------------------------------------------------------------
// Utility kernels
// ---------------------------------------------------------------------------
__global__ void k_zero_f32(float* __restrict__ p, long long n) {
    long long i = (long long)blockIdx.x * blockDim.x + threadIdx.x;
    long long st = (long long)gridDim.x * blockDim.x;
    for (; i < n; i += st) p[i] = 0.0f;
}
__global__ void k_fill_u32(unsigned* __restrict__ p, unsigned v, long long n) {
    long long i = (long long)blockIdx.x * blockDim.x + threadIdx.x;
    long long st = (long long)gridDim.x * blockDim.x;
    for (; i < n; i += st) p[i] = v;
}

// ---------------------------------------------------------------------------
// GAT attention pipeline
// ---------------------------------------------------------------------------
__global__ void k_attn_scores(const float* __restrict__ Hb,
                              const float* __restrict__ a_src,
                              const float* __restrict__ a_dst,
                              float* __restrict__ es, float* __restrict__ ed)
{
    int t = blockIdx.x * blockDim.x + threadIdx.x;     // (n, h)
    if (t >= N_NODES * NHEAD) return;
    int n = t / NHEAD, h = t % NHEAD;
    const float* hp = Hb + (long long)n * W3 + h * HID;
    const float* as = a_src + h * HID;
    const float* ad = a_dst + h * HID;
    float s = 0.f, d = 0.f;
#pragma unroll 4
    for (int k = 0; k < HID; ++k) { float v = hp[k]; s += v * as[k]; d += v * ad[k]; }
    es[t] = s; ed[t] = d;
}

// pass 1: e = leaky_relu(es[src]+ed[dst]); segment max into mkey[dst]
__global__ void k_edge_logits(const int* __restrict__ ei,
                              const float* __restrict__ es,
                              const float* __restrict__ ed,
                              float* __restrict__ ebuf,
                              unsigned* __restrict__ mkey)
{
    int e = blockIdx.x * blockDim.x + threadIdx.x;
    if (e >= ET_EDGES) return;
    int s, d; edge_sd(ei, e, s, d);
#pragma unroll
    for (int h = 0; h < NHEAD; ++h) {
        float v = es[s * NHEAD + h] + ed[d * NHEAD + h];
        v = (v > 0.f) ? v : NEG_SLOPE * v;
        ebuf[(long long)e * NHEAD + h] = v;
        atomicMax(&mkey[d * NHEAD + h], enc_f32(v));
    }
}

// pass 2: p = exp(e - m[dst]); segment sum into ssum[dst]
__global__ void k_edge_exp(const int* __restrict__ ei,
                           const unsigned* __restrict__ mkey,
                           float* __restrict__ ebuf,
                           float* __restrict__ ssum)
{
    int e = blockIdx.x * blockDim.x + threadIdx.x;
    if (e >= ET_EDGES) return;
    int s, d; edge_sd(ei, e, s, d);
#pragma unroll
    for (int h = 0; h < NHEAD; ++h) {
        float m = dec_f32(mkey[d * NHEAD + h]);
        float p = __expf(ebuf[(long long)e * NHEAD + h] - m);
        ebuf[(long long)e * NHEAD + h] = p;
        unsafeAtomicAdd(&ssum[d * NHEAD + h], p);
    }
}

// pass 3: alpha = p / s[dst]   (in place)
__global__ void k_edge_alpha(const int* __restrict__ ei,
                             const float* __restrict__ ssum,
                             float* __restrict__ ebuf)
{
    int e = blockIdx.x * blockDim.x + threadIdx.x;
    if (e >= ET_EDGES) return;
    int s, d; edge_sd(ei, e, s, d);
#pragma unroll
    for (int h = 0; h < NHEAD; ++h)
        ebuf[(long long)e * NHEAD + h] /= ssum[d * NHEAD + h];
}

// pass 4: out[dst, :] += alpha * h[src, :]   (one 384-thread block per edge)
__global__ __launch_bounds__(W3)
void k_edge_aggregate(const int* __restrict__ ei,
                      const float* __restrict__ ebuf,
                      const float* __restrict__ Hb,
                      float* __restrict__ Out)
{
    int e = blockIdx.x;
    int c = threadIdx.x;                 // 0..383
    int s, d; edge_sd(ei, e, s, d);
    float alpha = ebuf[(long long)e * NHEAD + (c >> 7)];
    unsafeAtomicAdd(&Out[(long long)d * W3 + c],
                    alpha * Hb[(long long)s * W3 + c]);
}

// ---------------------------------------------------------------------------
// BatchNorm (conv bias folded into stats; it cancels through BN but kept for
// fidelity).  part[blk][0:384]=sum, [384:768]=sumsq over 256-row chunk.
// ---------------------------------------------------------------------------
#define BN_ROWS 256
__global__ __launch_bounds__(W3)
void k_bn_partial(const float* __restrict__ X, const float* __restrict__ bias,
                  float* __restrict__ part)
{
    int c  = threadIdx.x;
    int r0 = blockIdx.x * BN_ROWS;
    int r1 = r0 + BN_ROWS; if (r1 > N_NODES) r1 = N_NODES;
    float b = bias[c];
    float s = 0.f, q = 0.f;
    for (int r = r0; r < r1; ++r) {
        float v = X[(long long)r * W3 + c] + b;
        s += v; q += v * v;
    }
    part[(long long)blockIdx.x * (2 * W3) + c]      = s;
    part[(long long)blockIdx.x * (2 * W3) + W3 + c] = q;
}

__global__ __launch_bounds__(W3)
void k_bn_finalize(const float* __restrict__ part, int nblk,
                   float* __restrict__ stats)
{
    int c = threadIdx.x;
    float s = 0.f, q = 0.f;
    for (int b = 0; b < nblk; ++b) {
        s += part[(long long)b * (2 * W3) + c];
        q += part[(long long)b * (2 * W3) + W3 + c];
    }
    float mu  = s / (float)N_NODES;
    float var = q / (float)N_NODES - mu * mu;
    stats[c]      = mu;
    stats[W3 + c] = rsqrtf(var + 1e-5f);
}

__global__ void k_bn_apply(float* __restrict__ X, const float* __restrict__ bias,
                           const float* __restrict__ stats,
                           const float* __restrict__ gamma,
                           const float* __restrict__ beta)
{
    long long total = (long long)N_NODES * W3;
    long long i  = (long long)blockIdx.x * blockDim.x + threadIdx.x;
    long long st = (long long)gridDim.x * blockDim.x;
    for (; i < total; i += st) {
        int c = (int)(i % W3);
        float v = X[i] + bias[c];
        X[i] = (v - stats[c]) * stats[W3 + c] * gamma[c] + beta[c];
    }
}

// ---------------------------------------------------------------------------
// Global mean pool
// ---------------------------------------------------------------------------
__global__ void k_pool_add(const float* __restrict__ X, const int* __restrict__ batch,
                           float* __restrict__ pooled)
{
    long long total = (long long)N_NODES * W3;
    long long i  = (long long)blockIdx.x * blockDim.x + threadIdx.x;
    long long st = (long long)gridDim.x * blockDim.x;
    for (; i < total; i += st) {
        int n = (int)(i / W3), c = (int)(i % W3);
        unsafeAtomicAdd(&pooled[(long long)batch[n] * W3 + c], X[i]);
    }
}
__global__ void k_pool_cnt(const int* __restrict__ batch, float* __restrict__ cnt) {
    int n = blockIdx.x * blockDim.x + threadIdx.x;
    if (n < N_NODES) unsafeAtomicAdd(&cnt[batch[n]], 1.0f);
}
__global__ void k_pool_div(float* __restrict__ pooled, const float* __restrict__ cnt) {
    int i = blockIdx.x * blockDim.x + threadIdx.x;
    if (i >= G_GRAPHS * W3) return;
    float c = cnt[i / W3];
    pooled[i] /= fmaxf(c, 1.0f);
}

// ---------------------------------------------------------------------------
// Head:  logits = gm @ lin4_w + b;  out = log_softmax(logits)
// ---------------------------------------------------------------------------
__global__ void k_head(const float* __restrict__ gm,
                       const float* __restrict__ w4, const float* __restrict__ b4,
                       float* __restrict__ out)
{
    int g = blockIdx.x * blockDim.x + threadIdx.x;
    if (g >= G_GRAPHS) return;
    float logit[NCLS];
#pragma unroll
    for (int j = 0; j < NCLS; ++j) logit[j] = b4[j];
    for (int k = 0; k < HID; ++k) {
        float v = gm[(long long)g * HID + k];
#pragma unroll
        for (int j = 0; j < NCLS; ++j) logit[j] += v * w4[k * NCLS + j];
    }
    float mx = logit[0];
#pragma unroll
    for (int j = 1; j < NCLS; ++j) mx = fmaxf(mx, logit[j]);
    float se = 0.f;
#pragma unroll
    for (int j = 0; j < NCLS; ++j) se += __expf(logit[j] - mx);
    float lse = mx + __logf(se);
#pragma unroll
    for (int j = 0; j < NCLS; ++j) out[(long long)g * NCLS + j] = logit[j] - lse;
}

// ---------------------------------------------------------------------------
// Host orchestration
// ---------------------------------------------------------------------------
extern "C" void kernel_launch(void* const* d_in, const int* in_sizes, int n_in,
                              void* d_out, int out_size, void* d_ws, size_t ws_size,
                              hipStream_t stream)
{
    const float* x       = (const float*)d_in[0];
    const int*   ei      = (const int*)  d_in[1];
    const int*   batch   = (const int*)  d_in[2];
    const float* lin1_w  = (const float*)d_in[3];
    const float* lin1_b  = (const float*)d_in[4];
    const float* lin2_w  = (const float*)d_in[5];
    const float* lin2_b  = (const float*)d_in[6];
    const float* w0      = (const float*)d_in[7];
    const float* as0     = (const float*)d_in[8];
    const float* ad0     = (const float*)d_in[9];
    const float* b0      = (const float*)d_in[10];
    const float* wl      = (const float*)d_in[11];
    const float* asl     = (const float*)d_in[12];
    const float* adl     = (const float*)d_in[13];
    const float* bl      = (const float*)d_in[14];
    const float* gamma   = (const float*)d_in[15];
    const float* beta    = (const float*)d_in[16];
    const float* lin3_w  = (const float*)d_in[17];
    const float* lin3_b  = (const float*)d_in[18];
    const float* lin4_w  = (const float*)d_in[19];
    const float* lin4_b  = (const float*)d_in[20];
    float* out = (float*)d_out;

    // ---- workspace carving (float units, 32B aligned chunks) ----
    const int bn_nblk = (N_NODES + BN_ROWS - 1) / BN_ROWS;   // 391
    float* ws = (float*)d_ws;
    size_t off = 0;
    auto alloc = [&](size_t nfloats) {
        float* p = ws + off;
        off += (nfloats + 7) & ~(size_t)7;      // keep 32-byte alignment
        return p;
    };
    auto alloc_bf = [&](size_t nbf16) { return (__bf16*)alloc((nbf16 + 1) / 2); };

    float*    Xcur  = alloc((size_t)N_NODES * W3);   // layer activations
    float*    Hb    = alloc((size_t)N_NODES * W3);   // projection output / stem tmp
    float*    es    = alloc((size_t)N_NODES * NHEAD);
    float*    ed    = alloc((size_t)N_NODES * NHEAD);
    unsigned* mkey  = (unsigned*)alloc((size_t)N_NODES * NHEAD);
    float*    ssum  = alloc((size_t)N_NODES * NHEAD);
    float*    ebuf  = alloc((size_t)ET_EDGES * NHEAD);
    float*    part  = alloc((size_t)bn_nblk * 2 * W3);
    float*    stats = alloc(2 * W3);
    float*    pooled= alloc((size_t)G_GRAPHS * W3);
    float*    cnt   = alloc(G_GRAPHS);
    float*    gm    = alloc((size_t)G_GRAPHS * HID);
    // packed bf16 weights (fragment-order)
    __bf16*   wp1   = alloc_bf((size_t)F_IN * 2 * HID);        // 32x256
    __bf16*   wp2   = alloc_bf((size_t)2 * HID * HID);         // 256x128
    __bf16*   wp0   = alloc_bf((size_t)HID * W3);              // 128x384
    __bf16*   wpl   = alloc_bf((size_t)(NLAYER - 1) * W3 * W3);// 3x 384x384
    __bf16*   wp3   = alloc_bf((size_t)W3 * HID);              // 384x128
    (void)ws_size; (void)in_sizes; (void)n_in; (void)out_size;

    const long long nW3 = (long long)N_NODES * W3;
    const int T = 256;
    auto blks = [](long long n, int t) { return (unsigned)((n + t - 1) / t); };
    dim3 gemm_blk(T);

    // ---- pack all weight matrices to bf16 fragment layout (tiny) ----
    k_pack_w<<<blks(F_IN * 2 * HID, T), T, 0, stream>>>(lin1_w, wp1, F_IN, 2 * HID);
    k_pack_w<<<blks(2 * HID * HID, T), T, 0, stream>>>(lin2_w, wp2, 2 * HID, HID);
    k_pack_w<<<blks(HID * W3, T), T, 0, stream>>>(w0, wp0, HID, W3);
    for (int i = 0; i < NLAYER - 1; ++i)
        k_pack_w<<<blks((long long)W3 * W3, T), T, 0, stream>>>(
            wl + (size_t)i * W3 * W3, wpl + (size_t)i * W3 * W3, W3, W3);
    k_pack_w<<<blks(W3 * HID, T), T, 0, stream>>>(lin3_w, wp3, W3, HID);

    // ---- MLP stem (WMMA bf16) ----
    {   // x[N,32] @ lin1[32,256] + b, relu  -> Hb (N x 256)
        dim3 g((N_NODES + MT - 1) / MT, (2 * HID) / NT);
        k_gemm_bf16<<<g, gemm_blk, 0, stream>>>(x, wp1, lin1_b, Hb,
                                                N_NODES, F_IN, 2 * HID, 1);
    }
    {   // Hb[N,256] @ lin2[256,128] + b, relu -> Xcur (N x 128)
        dim3 g((N_NODES + MT - 1) / MT, HID / NT);
        k_gemm_bf16<<<g, gemm_blk, 0, stream>>>(Hb, wp2, lin2_b, Xcur,
                                                N_NODES, 2 * HID, HID, 1);
    }

    // ---- GAT layers ----
    for (int layer = 0; layer < NLAYER; ++layer) {
        const int     Fin  = (layer == 0) ? HID : W3;
        const __bf16* Wp   = (layer == 0) ? wp0 : wpl + (size_t)(layer - 1) * W3 * W3;
        const float*  a_s  = (layer == 0) ? as0 : asl + (size_t)(layer - 1) * NHEAD * HID;
        const float*  a_d  = (layer == 0) ? ad0 : adl + (size_t)(layer - 1) * NHEAD * HID;
        const float*  bias = (layer == 0) ? b0  : bl  + (size_t)(layer - 1) * W3;

        // h = Xcur @ W   (WMMA)
        dim3 g((N_NODES + MT - 1) / MT, W3 / NT);
        k_gemm_bf16<<<g, gemm_blk, 0, stream>>>(Xcur, Wp, nullptr, Hb,
                                                N_NODES, Fin, W3, 0);

        // attention logits per node
        k_attn_scores<<<blks((long long)N_NODES * NHEAD, T), T, 0, stream>>>(
            Hb, a_s, a_d, es, ed);

        // init segment max / sum
        k_fill_u32<<<blks((long long)N_NODES * NHEAD, T), T, 0, stream>>>(
            mkey, NEG_INF_KEY, (long long)N_NODES * NHEAD);
        k_zero_f32<<<blks((long long)N_NODES * NHEAD, T), T, 0, stream>>>(
            ssum, (long long)N_NODES * NHEAD);

        // edge softmax
        k_edge_logits<<<blks(ET_EDGES, T), T, 0, stream>>>(ei, es, ed, ebuf, mkey);
        k_edge_exp   <<<blks(ET_EDGES, T), T, 0, stream>>>(ei, mkey, ebuf, ssum);
        k_edge_alpha <<<blks(ET_EDGES, T), T, 0, stream>>>(ei, ssum, ebuf);

        // aggregate messages into Xcur
        k_zero_f32<<<blks(nW3, T), T, 0, stream>>>(Xcur, nW3);
        k_edge_aggregate<<<ET_EDGES, W3, 0, stream>>>(ei, ebuf, Hb, Xcur);

        // BatchNorm (conv bias folded in), in place on Xcur
        k_bn_partial <<<bn_nblk, W3, 0, stream>>>(Xcur, bias, part);
        k_bn_finalize<<<1, W3, 0, stream>>>(part, bn_nblk, stats);
        k_bn_apply   <<<blks(nW3, T), T, 0, stream>>>(
            Xcur, bias, stats, gamma + (size_t)layer * W3, beta + (size_t)layer * W3);
    }

    // ---- global mean pool ----
    k_zero_f32<<<blks((long long)G_GRAPHS * W3, T), T, 0, stream>>>(
        pooled, (long long)G_GRAPHS * W3);
    k_zero_f32<<<blks(G_GRAPHS, T), T, 0, stream>>>(cnt, G_GRAPHS);
    k_pool_add<<<blks(nW3, T), T, 0, stream>>>(Xcur, batch, pooled);
    k_pool_cnt<<<blks(N_NODES, T), T, 0, stream>>>(batch, cnt);
    k_pool_div<<<blks((long long)G_GRAPHS * W3, T), T, 0, stream>>>(pooled, cnt);

    // ---- head ----
    {   // pooled[G,384] @ lin3[384,128] + b, relu -> gm
        dim3 g((G_GRAPHS + MT - 1) / MT, HID / NT);
        k_gemm_bf16<<<g, gemm_blk, 0, stream>>>(pooled, wp3, lin3_b, gm,
                                                G_GRAPHS, W3, HID, 1);
    }
    k_head<<<blks(G_GRAPHS, T), T, 0, stream>>>(gm, lin4_w, lin4_b, out);
}
